// Attention_25220047962551
// MI455X (gfx1250) — matrix operand order repaired and checked by reading, not compiled, not run on hardware
//
#include <hip/hip_runtime.h>
#include <math.h>
#include <stdint.h>

// ---------------------------------------------------------------------------
// Problem constants (match reference)
// ---------------------------------------------------------------------------
#define DIMV      4096
#define SEQ       2048
#define BSZV      2
#define NHEADS    32
#define NKVHEADS  8
#define HEADDIM   128
#define KVDIM     (NKVHEADS * HEADDIM)   // 1024
#define TOKENS    (BSZV * SEQ)           // 4096

typedef __attribute__((ext_vector_type(16))) __bf16 v16bf;
typedef __attribute__((ext_vector_type(8)))  float  v8f;

union V16 {
    v16bf b;
    uint4 q[2];
    unsigned short s[16];
};

__device__ __forceinline__ unsigned short f32_to_bf16(float f) {
    unsigned u = __float_as_uint(f);
    unsigned r = u + 0x7FFFu + ((u >> 16) & 1u);   // round-to-nearest-even
    return (unsigned short)(r >> 16);
}
__device__ __forceinline__ float bf16_to_f32(unsigned short h) {
    return __uint_as_float(((unsigned)h) << 16);
}

// Async global -> LDS copy of 16 bytes per lane (CDNA5 ASYNCcnt path).
// Ordering is enforced by the waits + workgroup barriers.
__device__ __forceinline__ void async_copy_b128(unsigned lds_byte_addr,
                                                const void* gaddr) {
    asm volatile("global_load_async_to_lds_b128 %0, %1, off"
                 :: "v"(lds_byte_addr), "v"(gaddr));
}
__device__ __forceinline__ void async_wait_all() {
    asm volatile("s_wait_asynccnt 0x0" ::: "memory");
}
__device__ __forceinline__ void async_wait_le4() {
    asm volatile("s_wait_asynccnt 0x4" ::: "memory");
}

// ---------------------------------------------------------------------------
// fp32 -> bf16 bulk convert
// ---------------------------------------------------------------------------
__global__ void cvt_f32_to_bf16(const float* __restrict__ in,
                                unsigned short* __restrict__ out, size_t n) {
    size_t i = (size_t)blockIdx.x * blockDim.x + threadIdx.x;
    if (i < n) out[i] = f32_to_bf16(in[i]);
}

// ---------------------------------------------------------------------------
// WMMA GEMM:  C[M,N] = A[M,K] * B[N,K]^T, bf16 in, f32 accumulate.
// Block = 256 threads (8 wave32). Workgroup tile 128(M) x 128(N), K step 32.
// Double-buffered LDS staging via GLOBAL_LOAD_ASYNC_TO_LDS_B128: the next
// K-slab's copies are in flight while the current slab feeds 8 WMMAs/wave.
// Async loads complete in order, so s_wait_asynccnt<=4 retires the older
// 4-copy group while the newer one stays outstanding.
// ---------------------------------------------------------------------------
template <bool OUT_BF16>
__global__ __launch_bounds__(256) void wmma_gemm_bf16(
    const unsigned short* __restrict__ A,   // [M][K] bf16, row major
    const unsigned short* __restrict__ B,   // [N][K] bf16, row major (i.e. B^T)
    void* __restrict__ Cv, int M, int N, int K) {
    __shared__ unsigned short As[2][128 * 32];  // 2 x 8 KB
    __shared__ unsigned short Bs[2][128 * 32];  // 2 x 8 KB

    const int m0   = blockIdx.y * 128;
    const int n0   = blockIdx.x * 128;
    const int t    = threadIdx.x;
    const int lane = t & 31;
    const int wave = t >> 5;                 // 0..7
    const int wm   = (wave >> 1) * 32;       // 0,32,64,96
    const int wn   = (wave & 1) * 64;        // 0,64
    const int hl   = lane >> 4;              // half-wave select
    const int l16  = lane & 15;

    // Staging assignment: each thread moves 32 contiguous bytes of A and of B.
    const int srow = t >> 1;                 // 0..127
    const int sseg = (t & 1) * 16;           // 0 or 16 (bf16 elements)

    const unsigned ldsA[2] = {
        (unsigned)(uintptr_t)(&As[0][srow * 32 + sseg]),
        (unsigned)(uintptr_t)(&As[1][srow * 32 + sseg])};
    const unsigned ldsB[2] = {
        (unsigned)(uintptr_t)(&Bs[0][srow * 32 + sseg]),
        (unsigned)(uintptr_t)(&Bs[1][srow * 32 + sseg])};
    const unsigned short* gA = A + (size_t)(m0 + srow) * K + sseg;
    const unsigned short* gB = B + (size_t)(n0 + srow) * K + sseg;

    v8f acc[2][4];
#pragma unroll
    for (int i = 0; i < 2; ++i)
#pragma unroll
        for (int j = 0; j < 4; ++j)
#pragma unroll
            for (int r = 0; r < 8; ++r) acc[i][j][r] = 0.0f;

    // Prologue: stage first K-slab into buffer 0.
    async_copy_b128(ldsA[0],      gA);
    async_copy_b128(ldsA[0] + 16, gA + 8);
    async_copy_b128(ldsB[0],      gB);
    async_copy_b128(ldsB[0] + 16, gB + 8);

    int buf = 0;
    for (int k0 = 0; k0 < K; k0 += 32) {
        const bool more = (k0 + 32) < K;
        if (more) {                       // stage next slab into other buffer
            const int nb = buf ^ 1;
            async_copy_b128(ldsA[nb],      gA + k0 + 32);
            async_copy_b128(ldsA[nb] + 16, gA + k0 + 40);
            async_copy_b128(ldsB[nb],      gB + k0 + 32);
            async_copy_b128(ldsB[nb] + 16, gB + k0 + 40);
            async_wait_le4();             // current slab's 4 copies retired
        } else {
            async_wait_all();
        }
        __syncthreads();

        // Fragments: A lane m = l16, K pairs hl*8..+7 / 16+hl*8..+7;
        //            B lane n = l16, K pairs hl*8..+7 / 16+hl*8..+7.
        const unsigned short* Ab = As[buf];
        const unsigned short* Bb = Bs[buf];
        V16 af[2], bfr[4];
#pragma unroll
        for (int i = 0; i < 2; ++i) {
            const unsigned short* p = &Ab[(wm + i * 16 + l16) * 32];
            af[i].q[0] = *(const uint4*)(p + hl * 8);
            af[i].q[1] = *(const uint4*)(p + 16 + hl * 8);
        }
#pragma unroll
        for (int j = 0; j < 4; ++j) {
            const unsigned short* p = &Bb[(wn + j * 16 + l16) * 32];
            bfr[j].q[0] = *(const uint4*)(p + hl * 8);
            bfr[j].q[1] = *(const uint4*)(p + 16 + hl * 8);
        }

#pragma unroll
        for (int i = 0; i < 2; ++i)
#pragma unroll
            for (int j = 0; j < 4; ++j)
                acc[i][j] = __builtin_amdgcn_wmma_f32_16x16x32_bf16(
                    false, af[i].b, false, bfr[j].b, (short)0, acc[i][j],
                    false, false);
        __syncthreads();                  // all waves done reading `buf`
        buf ^= 1;
    }

    // C/D layout: lane l, reg r -> m = r + 8*(l>=16), n = l&15
    const size_t base = (size_t)(m0 + wm + hl * 8) * N + (n0 + wn + l16);
#pragma unroll
    for (int i = 0; i < 2; ++i) {
#pragma unroll
        for (int j = 0; j < 4; ++j) {
#pragma unroll
            for (int r = 0; r < 8; ++r) {
                size_t addr = base + (size_t)(i * 16 + r) * N + j * 16;
                float v = acc[i][j][r];
                if (OUT_BF16)
                    ((unsigned short*)Cv)[addr] = f32_to_bf16(v);
                else
                    ((float*)Cv)[addr] = v;
            }
        }
    }
}

// ---------------------------------------------------------------------------
// RoPE, in place on a bf16 activation tensor [B*S][nheads*128].
// One thread per rotated pair.
// ---------------------------------------------------------------------------
__global__ void rope_bf16(unsigned short* __restrict__ q,
                          const float* __restrict__ cosb,
                          const float* __restrict__ sinb,
                          int nheads, int rowstride) {
    int idx = blockIdx.x * blockDim.x + threadIdx.x;
    int total = BSZV * SEQ * nheads * (HEADDIM / 2);
    if (idx >= total) return;
    int i = idx & 63;                       // pair index within head
    int h = (idx >> 6) % nheads;
    int s = (idx / (64 * nheads)) % SEQ;
    int b = idx / (64 * nheads * SEQ);
    size_t base = ((size_t)(b * SEQ + s)) * rowstride + h * HEADDIM + 2 * i;
    float c  = cosb[s * 64 + i];
    float sn = sinb[s * 64 + i];
    float a  = bf16_to_f32(q[base]);
    float bb = bf16_to_f32(q[base + 1]);
    q[base]     = f32_to_bf16(a * c - bb * sn);
    q[base + 1] = f32_to_bf16(a * sn + bb * c);
}

// ---------------------------------------------------------------------------
// Flash attention (causal, GQA). One wave32 per (batch, head, 16-query tile).
// All constant-trip loops force-unrolled so every vector-register index is a
// compile-time constant (no M0/movrel indexed-register access).
// ---------------------------------------------------------------------------
__global__ __launch_bounds__(32) void flash_attn_bf16(
    const unsigned short* __restrict__ Q,    // [B*S][4096]
    const unsigned short* __restrict__ Kb,   // [B*S][1024] (post-RoPE)
    const unsigned short* __restrict__ Vb,   // [B*S][1024]
    unsigned short* __restrict__ O) {        // [B*S][4096]
    const int tile = blockIdx.x;             // b*NHEADS*(SEQ/16) + h*(SEQ/16) + qt
    const int qt = tile & (SEQ / 16 - 1);
    const int h  = (tile >> 7) & (NHEADS - 1);
    const int b  = tile >> 12;
    const int q0 = qt * 16;

    const int lane = threadIdx.x;
    const int hl   = lane >> 4;
    const int l16  = lane & 15;

    __shared__ unsigned short Pt[16 * 32];       // probs tile, bf16
    __shared__ unsigned short Vt[32 * HEADDIM];  // V tile, bf16

    const unsigned short* qbase = Q  + (size_t)(b * SEQ) * DIMV + h * HEADDIM;
    const unsigned short* kbase = Kb + (size_t)(b * SEQ) * KVDIM + (h >> 2) * HEADDIM;
    const unsigned short* vbase = Vb + (size_t)(b * SEQ) * KVDIM + (h >> 2) * HEADDIM;

    // Q fragments: 16 rows x 128 depth = 4 x (16x32) A fragments.
    V16 qa[4];
#pragma unroll
    for (int d = 0; d < 4; ++d) {
        const unsigned short* p = qbase + (size_t)(q0 + l16) * DIMV + d * 32;
        qa[d].q[0] = *(const uint4*)(p + hl * 8);
        qa[d].q[1] = *(const uint4*)(p + 16 + hl * 8);
    }

    v8f   o[8];
    float mrow[8], lrow[8];
#pragma unroll
    for (int j = 0; j < 8; ++j)
#pragma unroll
        for (int r = 0; r < 8; ++r) o[j][r] = 0.0f;
#pragma unroll
    for (int r = 0; r < 8; ++r) { mrow[r] = -__builtin_inff(); lrow[r] = 0.0f; }

    const float scale = 0.08838834764831845f;    // 1/sqrt(128)
    const unsigned ldsV = (unsigned)(uintptr_t)(&Vt[lane * HEADDIM]);

    for (int k0 = 0; k0 <= q0 + 15; k0 += 32) {
        // ---- kick off async V tile staging (32 keys x 128 dims, row/lane) ----
        {
            const unsigned short* gv = vbase + (size_t)(k0 + lane) * KVDIM;
#pragma unroll
            for (int c4 = 0; c4 < 16; ++c4)
                async_copy_b128(ldsV + c4 * 16, gv + c4 * 8);
        }

        // ---- S = Q * K^T for 32 keys (two 16-key n-halves) ----
        // Load both n-half fragments per depth step before the two WMMAs so
        // the second WMMA issues without an extra load-wait.
        v8f ct0 = {}, ct1 = {};
#pragma unroll
        for (int d = 0; d < 4; ++d) {
            const unsigned short* p0 =
                kbase + (size_t)(k0 + l16) * KVDIM + d * 32;
            const unsigned short* p1 = p0 + (size_t)16 * KVDIM;
            V16 kf0, kf1;
            kf0.q[0] = *(const uint4*)(p0 + hl * 8);
            kf0.q[1] = *(const uint4*)(p0 + 16 + hl * 8);
            kf1.q[0] = *(const uint4*)(p1 + hl * 8);
            kf1.q[1] = *(const uint4*)(p1 + 16 + hl * 8);
            ct0 = __builtin_amdgcn_wmma_f32_16x16x32_bf16(
                false, qa[d].b, false, kf0.b, (short)0, ct0, false, false);
            ct1 = __builtin_amdgcn_wmma_f32_16x16x32_bf16(
                false, qa[d].b, false, kf1.b, (short)0, ct1, false, false);
        }

        // ---- scale + causal mask + online softmax (per-row, wave32 shuffles) ----
#pragma unroll
        for (int r = 0; r < 8; ++r) {
            int m = q0 + r + hl * 8;
            float s0 = ct0[r] * scale; if (k0 + l16      > m) s0 = -__builtin_inff();
            float s1 = ct1[r] * scale; if (k0 + 16 + l16 > m) s1 = -__builtin_inff();
            float mx = fmaxf(s0, s1);
#pragma unroll
            for (int off = 8; off >= 1; off >>= 1)
                mx = fmaxf(mx, __shfl_xor(mx, off, 32));
            float mnew  = fmaxf(mrow[r], mx);
            float alpha = __expf(mrow[r] - mnew);
            float p0 = __expf(s0 - mnew);
            float p1 = __expf(s1 - mnew);
            float rs = p0 + p1;
#pragma unroll
            for (int off = 8; off >= 1; off >>= 1)
                rs += __shfl_xor(rs, off, 32);
            lrow[r] = lrow[r] * alpha + rs;
            mrow[r] = mnew;
            int ml = r + hl * 8;
            Pt[ml * 32 + l16]      = f32_to_bf16(p0);
            Pt[ml * 32 + 16 + l16] = f32_to_bf16(p1);
#pragma unroll
            for (int j = 0; j < 8; ++j) o[j][r] *= alpha;
        }

        async_wait_all();      // V tile resident in LDS
        __syncthreads();

        // ---- P as A-fragment (16x32 over keys) ----
        V16 pa;
        pa.q[0] = *(const uint4*)(&Pt[l16 * 32 + hl * 8]);
        pa.q[1] = *(const uint4*)(&Pt[l16 * 32 + 16 + hl * 8]);

        // ---- O += P * V, 8 dim-chunks of 16 ----
#pragma unroll
        for (int j = 0; j < 8; ++j) {
            V16 vf;   // B fragment: lane -> dim column, depth = keys
#pragma unroll
            for (int v = 0; v < 8; ++v) {
                int k = (v < 4) ? (hl * 8 + 2 * v) : (16 + hl * 8 + 2 * (v - 4));
                vf.s[2 * v]     = Vt[k * HEADDIM + j * 16 + l16];
                vf.s[2 * v + 1] = Vt[(k + 1) * HEADDIM + j * 16 + l16];
            }
            o[j] = __builtin_amdgcn_wmma_f32_16x16x32_bf16(
                false, pa.b, false, vf.b, (short)0, o[j], false, false);
        }
        __syncthreads();
    }

    // ---- normalize and emit bf16 ----
#pragma unroll
    for (int r = 0; r < 8; ++r) {
        float inv = 1.0f / lrow[r];
        int m = q0 + r + hl * 8;
        unsigned short* orow = O + (size_t)(b * SEQ + m) * DIMV + h * HEADDIM;
#pragma unroll
        for (int j = 0; j < 8; ++j)
            orow[j * 16 + l16] = f32_to_bf16(o[j][r] * inv);
    }
}

// ---------------------------------------------------------------------------
// Host-side orchestration
// ---------------------------------------------------------------------------
extern "C" void kernel_launch(void* const* d_in, const int* in_sizes, int n_in,
                              void* d_out, int out_size, void* d_ws, size_t ws_size,
                              hipStream_t stream) {
    (void)in_sizes; (void)n_in; (void)out_size; (void)ws_size;
    const float* x    = (const float*)d_in[0];
    const float* wq   = (const float*)d_in[1];
    const float* wk   = (const float*)d_in[2];
    const float* wv   = (const float*)d_in[3];
    const float* wo   = (const float*)d_in[4];
    const float* fcos = (const float*)d_in[5];
    const float* fsin = (const float*)d_in[6];
    float* out = (float*)d_out;

    char* ws = (char*)d_ws;
    const size_t MB = 1024 * 1024;
    unsigned short* xbf  = (unsigned short*)(ws +   0 * MB);  // 32 MB
    unsigned short* wqbf = (unsigned short*)(ws +  32 * MB);  // 32 MB
    unsigned short* wkbf = (unsigned short*)(ws +  64 * MB);  //  8 MB
    unsigned short* wvbf = (unsigned short*)(ws +  72 * MB);  //  8 MB
    unsigned short* wobf = (unsigned short*)(ws +  80 * MB);  // 32 MB
    unsigned short* qbf  = (unsigned short*)(ws + 112 * MB);  // 32 MB
    unsigned short* kbf  = (unsigned short*)(ws + 144 * MB);  //  8 MB
    unsigned short* vbf  = (unsigned short*)(ws + 152 * MB);  //  8 MB
    unsigned short* abf  = (unsigned short*)(ws + 160 * MB);  // 32 MB

    const size_t nx  = (size_t)TOKENS * DIMV;
    const size_t nwq = (size_t)DIMV * DIMV;
    const size_t nwk = (size_t)KVDIM * DIMV;

    cvt_f32_to_bf16<<<dim3((unsigned)((nx  + 255) / 256)), 256, 0, stream>>>(x,  xbf,  nx);
    cvt_f32_to_bf16<<<dim3((unsigned)((nwq + 255) / 256)), 256, 0, stream>>>(wq, wqbf, nwq);
    cvt_f32_to_bf16<<<dim3((unsigned)((nwk + 255) / 256)), 256, 0, stream>>>(wk, wkbf, nwk);
    cvt_f32_to_bf16<<<dim3((unsigned)((nwk + 255) / 256)), 256, 0, stream>>>(wv, wvbf, nwk);
    cvt_f32_to_bf16<<<dim3((unsigned)((nwq + 255) / 256)), 256, 0, stream>>>(wo, wobf, nwq);

    // Projections: xq = x @ wq^T etc., bf16 out (workgroup tile 128x128)
    wmma_gemm_bf16<true><<<dim3(DIMV / 128,  TOKENS / 128), 256, 0, stream>>>(
        xbf, wqbf, qbf, TOKENS, DIMV, DIMV);
    wmma_gemm_bf16<true><<<dim3(KVDIM / 128, TOKENS / 128), 256, 0, stream>>>(
        xbf, wkbf, kbf, TOKENS, KVDIM, DIMV);
    wmma_gemm_bf16<true><<<dim3(KVDIM / 128, TOKENS / 128), 256, 0, stream>>>(
        xbf, wvbf, vbf, TOKENS, KVDIM, DIMV);

    // RoPE in place on Q and K
    {
        int nq = BSZV * SEQ * NHEADS * (HEADDIM / 2);
        int nk = BSZV * SEQ * NKVHEADS * (HEADDIM / 2);
        rope_bf16<<<(nq + 255) / 256, 256, 0, stream>>>(qbf, fcos, fsin, NHEADS,  DIMV);
        rope_bf16<<<(nk + 255) / 256, 256, 0, stream>>>(kbf, fcos, fsin, NKVHEADS, KVDIM);
    }

    // Causal GQA flash attention: one wave per (b, h, 16-row query tile)
    flash_attn_bf16<<<BSZV * NHEADS * (SEQ / 16), 32, 0, stream>>>(qbf, kbf, vbf, abf);

    // Output projection: out = attn @ wo^T, f32 out
    wmma_gemm_bf16<false><<<dim3(DIMV / 128, TOKENS / 128), 256, 0, stream>>>(
        abf, wobf, (void*)out, TOKENS, DIMV, DIMV);
}
